// Linear_43026982371802
// MI455X (gfx1250) — matrix operand order repaired and checked by reading, not compiled, not run on hardware
//
#include <hip/hip_runtime.h>
#include <stdint.h>

typedef __attribute__((ext_vector_type(16))) _Float16 v16h;
typedef __attribute__((ext_vector_type(8)))  float    v8f;
typedef __attribute__((ext_vector_type(4)))  int      v4i;
typedef __attribute__((ext_vector_type(4)))  uint32_t v4u;

#define IN_F  4096
#define OUT_F 16384

// Reinterpret 8 packed dwords as a 16-half WMMA fragment.
union HUnion {
  v16h     h;
  uint32_t u[8];
  v4u      q[2];
};

// Two fp10 (exp:8, frac:2) elements -> one dword holding two fp16 bit patterns.
// fp16 bits = (exp << 8) | (frac << 6); low 6 mantissa bits zero.
__device__ __forceinline__ uint32_t pack_fp10_pair(int e0, int f0, int e1, int f1) {
  return ((uint32_t)e0 << 8)  | ((uint32_t)f0 << 6) |
         ((uint32_t)e1 << 24) | ((uint32_t)f1 << 22);
}

// One K=32 pipeline stage: raw weight exp/frac dwords + raw A-fragment dwords.
struct Stage {
  v4i e[4];   // 16 exp int32 (this lane's B column, 16 consecutive K)
  v4i f[4];   // 16 frac int32
  v4u a[8];   // 4 A fragments x 2 b128 chunks
};

__device__ __forceinline__ void stage_load(Stage& s,
                                           const int* __restrict__ eRow,
                                           const int* __restrict__ fRow,
                                           const _Float16* __restrict__ x,
                                           int k, int aRow, int aKOff, int bKOff) {
  const int* ep = eRow + k + bKOff;
  const int* fp = fRow + k + bKOff;
#pragma unroll
  for (int i = 0; i < 4; ++i)
    s.e[i] = __builtin_nontemporal_load((const v4i*)(ep + 4 * i));
#pragma unroll
  for (int i = 0; i < 4; ++i)
    s.f[i] = __builtin_nontemporal_load((const v4i*)(fp + 4 * i));
#pragma unroll
  for (int t = 0; t < 4; ++t) {
    const _Float16* p = x + (size_t)(t * 16 + aRow) * IN_F + (k + aKOff);
    s.a[2 * t]     = *(const v4u*)(p);        // K = aKOff .. aKOff+7
    s.a[2 * t + 1] = *(const v4u*)(p + 8);    // K = aKOff+8 .. aKOff+15
  }
}

__device__ __forceinline__ void stage_compute(const Stage& s, v8f acc[4]) {
  // Dequantize B fragment (16 fp10 -> 8 packed fp16 dwords).
  HUnion b;
#pragma unroll
  for (int i = 0; i < 4; ++i) {
    b.u[2 * i]     = pack_fp10_pair(s.e[i][0], s.f[i][0], s.e[i][1], s.f[i][1]);
    b.u[2 * i + 1] = pack_fp10_pair(s.e[i][2], s.f[i][2], s.e[i][3], s.f[i][3]);
  }
#pragma unroll
  for (int t = 0; t < 4; ++t) {
    HUnion a;
    a.q[0] = s.a[2 * t];
    a.q[1] = s.a[2 * t + 1];
    acc[t] = __builtin_amdgcn_wmma_f32_16x16x32_f16(
        /*neg_a=*/false, a.h, /*neg_b=*/false, b.h,
        /*c_mod=*/(short)0, acc[t], /*reuse_a=*/false, /*reuse_b=*/false);
  }
}

__global__ __launch_bounds__(256)
void fp10_linear_wmma(const _Float16* __restrict__ x,       // [64, 4096] f16
                      const int*      __restrict__ w_exp,   // [16384, 4096] i32
                      const int*      __restrict__ w_frac,  // [16384, 4096] i32
                      const int*      __restrict__ b_exp,   // [16384] i32
                      const int*      __restrict__ b_frac,  // [16384] i32
                      float*          __restrict__ out) {   // [64, 16384] f32
  const int lane  = (int)(threadIdx.x & 31u);
  const int wave  = (int)(threadIdx.x >> 5u);
  const int nBase = (int)blockIdx.x * 128 + wave * 16;   // this wave's N tile
  const int col   = nBase + (lane & 15);                 // B column / output column
  const int hi16  = lane >> 4;                           // half-wave select

  const int aKOff = hi16 * 16;   // A: 16 consecutive K per half-wave
  const int bKOff = hi16 * 16;   // B: 16 consecutive K per half-wave
  const int aRow  = lane & 15;   // A: row M%16

  const int* __restrict__ eRow = w_exp  + (size_t)col * IN_F;
  const int* __restrict__ fRow = w_frac + (size_t)col * IN_F;

  // fp10 bias dequant hoisted so its loads overlap the main loop.
  uint16_t bb = (uint16_t)(((uint32_t)b_exp[col] << 8) | ((uint32_t)b_frac[col] << 6));
  _Float16 bh;
  __builtin_memcpy(&bh, &bb, sizeof(bh));
  const float bias = (float)bh;

  v8f acc[4] = {};   // 4 M-tiles (M = 0..15, 16..31, 32..47, 48..63) x one N tile

  // Software-pipelined double buffer: stage n+1's 16 loads are in flight while
  // stage n's dequant + 4 WMMAs execute -> partial loadcnt waits, not flushes.
  Stage s0, s1;
  stage_load(s0, eRow, fRow, x, 0, aRow, aKOff, bKOff);

  for (int k = 0; k < IN_F; k += 64) {
    stage_load(s1, eRow, fRow, x, k + 32, aRow, aKOff, bKOff);
    if (k + 96 < IN_F) {               // prime GL2 two stages ahead
      __builtin_prefetch(eRow + k + 96 + bKOff, 0, 1);
      __builtin_prefetch(fRow + k + 96 + bKOff, 0, 1);
    }
    stage_compute(s0, acc);

    if (k + 64 < IN_F)
      stage_load(s0, eRow, fRow, x, k + 64, aRow, aKOff, bKOff);
    if (k + 128 < IN_F) {
      __builtin_prefetch(eRow + k + 128 + bKOff, 0, 1);
      __builtin_prefetch(fRow + k + 128 + bKOff, 0, 1);
    }
    stage_compute(s1, acc);
  }

  // Epilogue: D layout — lanes 0-15: M = vgpr; lanes 16-31: M = 8 + vgpr.
#pragma unroll
  for (int t = 0; t < 4; ++t) {
    const int rowBase = t * 16 + hi16 * 8;
#pragma unroll
    for (int v = 0; v < 8; ++v) {
      out[(size_t)(rowBase + v) * OUT_F + col] = acc[t][v] + bias;
    }
  }
}

extern "C" void kernel_launch(void* const* d_in, const int* in_sizes, int n_in,
                              void* d_out, int out_size, void* d_ws, size_t ws_size,
                              hipStream_t stream) {
  const _Float16* x      = (const _Float16*)d_in[0];
  const int*      w_exp  = (const int*)d_in[1];
  const int*      w_frac = (const int*)d_in[2];
  const int*      b_exp  = (const int*)d_in[3];
  const int*      b_frac = (const int*)d_in[4];
  float*          out    = (float*)d_out;

  dim3 grid(OUT_F / 128);   // 128 blocks; each covers 8 waves x 16 columns
  dim3 block(256);          // 8 wave32 waves per block
  hipLaunchKernelGGL(fp10_linear_wmma, grid, block, 0, stream,
                     x, w_exp, w_frac, b_exp, b_frac, out);
}
